// VmambaIR_83915071029772
// MI455X (gfx1250) — matrix operand order repaired and checked by reading, not compiled, not run on hardware
//
#include <hip/hip_runtime.h>
#include <math.h>

// ---------------------------------------------------------------------------
// Types for WMMA
// ---------------------------------------------------------------------------
typedef __attribute__((ext_vector_type(16))) __bf16          v16bf;
typedef __attribute__((ext_vector_type(16))) unsigned short  v16u;
typedef __attribute__((ext_vector_type(8)))  float           v8f;

__device__ __forceinline__ unsigned short f2bf(float f) {
    unsigned u = __float_as_uint(f);
    unsigned r = u + 0x7FFFu + ((u >> 16) & 1u);   // round-to-nearest-even
    return (unsigned short)(r >> 16);
}
__device__ __forceinline__ float siluf(float v) { return v / (1.f + expf(-v)); }
__device__ __forceinline__ float softplusf(float v) { return (v > 20.f) ? v : log1pf(expf(v)); }
__device__ __forceinline__ float geluf(float v) { return 0.5f * v * (1.f + erff(v * 0.70710678118654752f)); }

__device__ __forceinline__ float epilogue(float v, const float* bias, int biasMode,
                                          int act, int m, int n) {
    if (biasMode == 1) v += bias[n];
    else if (biasMode == 2) v += bias[m];
    if (act == 1) v = siluf(v);
    else if (act == 2) v = softplusf(v);
    return v;
}

// ---------------------------------------------------------------------------
// WMMA GEMM:  C[m,n] = act( sum_k A(m,k) * B(k,n) + bias )
//   A(m,k) = A[m*lda+k]   (row-major always)
//   B(k,n) = TB ? W[k*ldb+n] : W[n*ldb+k]
//   biasMode: 0 none, 1 per-n, 2 per-m.  act: 0 none, 1 silu, 2 softplus.
//
//   Block = 4 waves, 64x32 C macro-tile: waves split M (4 x 16 rows), each
//   wave computes TWO 16x16 N-tiles reusing one A fragment across two
//   v_wmma_f32_16x16x32_bf16 ops. The 32x32 B tile is shared block-wide and
//   staged into LDS with gfx1250 async-to-LDS b128 loads (2 per thread; the
//   offset: immediate shifts BOTH global and LDS addresses), synchronized by
//   s_wait_asynccnt + barrier. Boundary path uses min()-clamped addresses so
//   loads stay unconditional (no divergent exec ladders around the WMMA).
// ---------------------------------------------------------------------------
template<int TB>
__global__ void k_gemm_t(const float* __restrict__ A, const float* __restrict__ Wp,
                         const float* __restrict__ bias, float* __restrict__ C,
                         int M, int N, int K, int lda, int ldb, int ldc,
                         int biasMode, int act, long sA, long sB, long sC) {
    // 32 rows x 36 floats (144B row stride, 16B aligned). TB==0: [n][kk],
    // TB==1: [kk][n].
    __shared__ float bt[32 * 36];

    A  += (long)blockIdx.z * sA;
    Wp += (long)blockIdx.z * sB;
    C  += (long)blockIdx.z * sC;

    const int tid  = threadIdx.x;
    const int wave = tid >> 5;
    const int lid  = tid & 31;
    const int m0   = (blockIdx.y * 4 + wave) * 16;
    const int n0   = blockIdx.x * 32;
    const int half = lid >> 4;
    const int nl   = lid & 15;
    const int mA   = m0 + nl;

    // cooperative staging map: 128 threads, 32 rows x 2 quads each
    const int stR = tid >> 2;            // row 0..31
    const int stQ = (tid & 3) << 3;      // col base 0,8,16,24 (two quads)

    v8f acc0 = {}, acc1 = {};
    for (int kb = 0; kb < K; kb += 32) {
        const bool kfull = (kb + 32 <= K);

        // ---------------- B fragments (two 16-wide N tiles) ----------------
        v16u bu0, bu1;
        if (kfull && (n0 + 32 <= N) && ((ldb & 3) == 0)) {
            const float* gp;
            if (TB == 0) gp = Wp + (long)(n0 + stR) * ldb + kb + stQ;   // row=n
            else         gp = Wp + (long)(kb + stR) * ldb + n0 + stQ;   // row=kk
            unsigned lo = (unsigned)(size_t)&bt[stR * 36 + stQ];
            asm volatile("global_load_async_to_lds_b128 %0, %1, off"
                         :: "v"(lo), "v"(gp) : "memory");
            asm volatile("global_load_async_to_lds_b128 %0, %1, off offset:16"
                         :: "v"(lo), "v"(gp) : "memory");
            asm volatile("s_wait_asynccnt 0" ::: "memory");
            __syncthreads();
            #pragma unroll
            for (int e = 0; e < 16; ++e) {
                int v = e >> 1, p = e & 1;
                int kk = half * 16 + 2 * v + p;
                float f0, f1;
                if (TB == 0) { f0 = bt[nl * 36 + kk];        f1 = bt[(16 + nl) * 36 + kk]; }
                else         { f0 = bt[kk * 36 + nl];        f1 = bt[kk * 36 + 16 + nl];  }
                bu0[e] = f2bf(f0);
                bu1[e] = f2bf(f1);
            }
            __syncthreads();
        } else {
            #pragma unroll
            for (int e = 0; e < 16; ++e) {
                int v = e >> 1, p = e & 1;
                int kk  = kb + half * 16 + 2 * v + p;
                int kc  = min(kk, K - 1);
                int n1  = n0 + nl, n2 = n0 + 16 + nl;
                int nc1 = min(n1, N - 1), nc2 = min(n2, N - 1);
                float f0 = (TB == 0) ? Wp[(long)nc1 * ldb + kc] : Wp[(long)kc * ldb + nc1];
                float f1 = (TB == 0) ? Wp[(long)nc2 * ldb + kc] : Wp[(long)kc * ldb + nc2];
                bool ok = (kk < K);
                bu0[e] = f2bf((ok && n1 < N) ? f0 : 0.f);
                bu1[e] = f2bf((ok && n2 < N) ? f1 : 0.f);
            }
        }

        // ---------------- A fragment ----------------
        v16u au;
        if (kfull && (m0 + 16 <= M) && ((lda & 1) == 0)) {
            const float* arow = A + (long)mA * lda + kb + half * 16;
            #pragma unroll
            for (int v = 0; v < 8; ++v) {
                int k0 = 2 * v;
                float2 f2 = *(const float2*)(arow + k0);
                au[2 * v]     = f2bf(f2.x);
                au[2 * v + 1] = f2bf(f2.y);
            }
        } else {
            #pragma unroll
            for (int e = 0; e < 16; ++e) {
                int v = e >> 1, p = e & 1;
                int ka = half * 16 + 2 * v + p;
                int k  = kb + ka;
                int kc = min(k, K - 1);
                int mc = min(mA, M - 1);
                float fa = A[(long)mc * lda + kc];
                au[e] = f2bf((k < K && mA < M) ? fa : 0.f);
            }
        }

        v16bf a  = __builtin_bit_cast(v16bf, au);
        v16bf b0 = __builtin_bit_cast(v16bf, bu0);
        v16bf b1 = __builtin_bit_cast(v16bf, bu1);
        acc0 = __builtin_amdgcn_wmma_f32_16x16x32_bf16(false, a, false, b0,
                                                       (short)0, acc0, false, false);
        acc1 = __builtin_amdgcn_wmma_f32_16x16x32_bf16(false, a, false, b1,
                                                       (short)0, acc1, false, false);
    }

    #pragma unroll
    for (int r = 0; r < 8; ++r) {
        int m = m0 + r + 8 * half;
        if (m < M) {
            int n1 = n0 + nl, n2 = n0 + 16 + nl;
            if (n1 < N) C[(long)m * ldc + n1] = epilogue(acc0[r], bias, biasMode, act, m, n1);
            if (n2 < N) C[(long)m * ldc + n2] = epilogue(acc1[r], bias, biasMode, act, m, n2);
        }
    }
}

// ---------------------------------------------------------------------------
// Direct conv kernels
// ---------------------------------------------------------------------------
__global__ void k_conv2d(const float* __restrict__ x, const float* __restrict__ w,
                         const float* __restrict__ bias, float* __restrict__ out,
                         int B, int Cin, int Hin, int Win, int Cout, int Hout, int Wout,
                         int KH, int KW, int stride, int pad) {
    long total = (long)B * Cout * Hout * Wout;
    long i = (long)blockIdx.x * blockDim.x + threadIdx.x;
    if (i >= total) return;
    int xo = (int)(i % Wout); long t = i / Wout;
    int yo = (int)(t % Hout); t /= Hout;
    int o  = (int)(t % Cout); int b = (int)(t / Cout);
    float acc = bias[o];
    for (int ci = 0; ci < Cin; ++ci)
        for (int ky = 0; ky < KH; ++ky) {
            int ys = yo * stride - pad + ky;
            if (ys < 0 || ys >= Hin) continue;
            for (int kx = 0; kx < KW; ++kx) {
                int xs = xo * stride - pad + kx;
                if (xs < 0 || xs >= Win) continue;
                acc += x[(((long)b * Cin + ci) * Hin + ys) * Win + xs] *
                       w[(((long)o * Cin + ci) * KH + ky) * KW + kx];
            }
        }
    out[i] = acc;
}

// Transposed conv 4x4 stride 2; w layout (Cin, Cout, 4, 4)
__global__ void k_convT(const float* __restrict__ x, const float* __restrict__ w,
                        const float* __restrict__ bias, float* __restrict__ out,
                        int B, int Cin, int Hin, int Win, int Cout) {
    int Hout = Hin * 2, Wout = Win * 2;
    long total = (long)B * Cout * Hout * Wout;
    long i = (long)blockIdx.x * blockDim.x + threadIdx.x;
    if (i >= total) return;
    int xq = (int)(i % Wout); long t = i / Wout;
    int yq = (int)(t % Hout); t /= Hout;
    int o  = (int)(t % Cout); int b = (int)(t / Cout);
    float acc = bias[o];
    for (int ci = 0; ci < Cin; ++ci)
        for (int ky = 0; ky < 4; ++ky) {
            int p = yq - 2 + ky;
            if (p < 0 || (p & 1)) continue;
            int ys = p >> 1; if (ys >= Hin) continue;
            for (int kx = 0; kx < 4; ++kx) {
                int q = xq - 2 + kx;
                if (q < 0 || (q & 1)) continue;
                int xs = q >> 1; if (xs >= Win) continue;
                acc += x[(((long)b * Cin + ci) * Hin + ys) * Win + xs] *
                       w[(((long)ci * Cout + o) * 4 + (3 - ky)) * 4 + (3 - kx)];
            }
        }
    out[i] = acc;
}

// Depthwise 3x3 pad 1. src may have a larger per-batch stride (channel slice).
__global__ void k_dwconv3(const float* __restrict__ x, const float* __restrict__ w,
                          const float* __restrict__ bias, float* __restrict__ out,
                          int B, int C, int H, int W, long srcBatchStride, int act) {
    long total = (long)B * C * H * W;
    long i = (long)blockIdx.x * blockDim.x + threadIdx.x;
    if (i >= total) return;
    int xq = (int)(i % W); long t = i / W;
    int yq = (int)(t % H); t /= H;
    int c  = (int)(t % C); int b = (int)(t / C);
    const float* src = x + (long)b * srcBatchStride + (long)c * H * W;
    float acc = bias[c];
    #pragma unroll
    for (int dy = 0; dy < 3; ++dy) {
        int ys = yq + dy - 1;
        if (ys < 0 || ys >= H) continue;
        #pragma unroll
        for (int dx = 0; dx < 3; ++dx) {
            int xs = xq + dx - 1;
            if (xs < 0 || xs >= W) continue;
            acc += src[(long)ys * W + xs] * w[c * 9 + dy * 3 + dx];
        }
    }
    if (act == 1) acc = siluf(acc);
    out[i] = acc;
}

// ---------------------------------------------------------------------------
// GroupNorm: one block per (group, batch)
// ---------------------------------------------------------------------------
__global__ void k_gn(const float* __restrict__ x, const float* __restrict__ sc,
                     const float* __restrict__ bi, float* __restrict__ out,
                     int C, int HW, int g) {
    int gi = blockIdx.x, b = blockIdx.y;
    int cpg = C / g;
    long base = ((long)b * C + (long)gi * cpg) * HW;
    long n = (long)cpg * HW;
    float s = 0.f, sq = 0.f;
    for (long i = threadIdx.x; i < n; i += blockDim.x) {
        float v = x[base + i]; s += v; sq += v * v;
    }
    __shared__ float s1[256], s2[256];
    s1[threadIdx.x] = s; s2[threadIdx.x] = sq;
    __syncthreads();
    for (int st = 128; st > 0; st >>= 1) {
        if ((int)threadIdx.x < st) { s1[threadIdx.x] += s1[threadIdx.x + st]; s2[threadIdx.x] += s2[threadIdx.x + st]; }
        __syncthreads();
    }
    float mean = s1[0] / (float)n;
    float var  = s2[0] / (float)n - mean * mean;
    float inv  = rsqrtf(var + 1e-5f);
    for (long i = threadIdx.x; i < n; i += blockDim.x) {
        int c = gi * cpg + (int)(i / HW);
        out[base + i] = (x[base + i] - mean) * inv * sc[c] + bi[c];
    }
}

// ---------------------------------------------------------------------------
// Mamba pieces
// ---------------------------------------------------------------------------
__global__ void k_cconv1d(const float* __restrict__ xz, const float* __restrict__ cw,
                          const float* __restrict__ cb, float* __restrict__ u,
                          int B, int L, int di) {
    long total = (long)B * L * di;
    long i = (long)blockIdx.x * blockDim.x + threadIdx.x;
    if (i >= total) return;
    int d = (int)(i % di); long t = i / di;
    int l = (int)(t % L); int b = (int)(t / L);
    int ld = 2 * di;
    const float* col = xz + (long)b * L * ld + d;
    float acc = cb[d];
    #pragma unroll
    for (int k = 0; k < 4; ++k) {
        int ls = l - 3 + k;
        if (ls >= 0) acc += cw[d * 4 + k] * col[(long)ls * ld];
    }
    u[i] = siluf(acc);
}

// selective scan: one thread per (b,d); 16 SSM states in registers.
__global__ void k_scan(const float* __restrict__ delta, const float* __restrict__ Bm,
                       const float* __restrict__ Cm, const float* __restrict__ u,
                       const float* __restrict__ z, const float* __restrict__ A_log,
                       const float* __restrict__ Dp, float* __restrict__ y,
                       int B, int L, int di, int ldx, int ldz) {
    int i = blockIdx.x * blockDim.x + threadIdx.x;
    if (i >= B * di) return;
    int d = i % di, b = i / di;
    float Av[16], h[16];
    #pragma unroll
    for (int n = 0; n < 16; ++n) { Av[n] = -expf(A_log[d * 16 + n]); h[n] = 0.f; }
    float Dv = Dp[d];
    for (int l = 0; l < L; ++l) {
        long t = (long)b * L + l;
        float dt = delta[t * di + d];
        float ut = u[t * di + d];
        const float* br = Bm + t * ldx;
        const float* cr = Cm + t * ldx;
        float du = dt * ut, s = 0.f;
        #pragma unroll
        for (int n = 0; n < 16; ++n) {
            h[n] = expf(dt * Av[n]) * h[n] + du * br[n];
            s += h[n] * cr[n];
        }
        float zv = z[t * ldz + d];
        y[t * di + d] = (s + ut * Dv) * siluf(zv);
    }
}

// ---------------------------------------------------------------------------
// Elementwise / glue kernels
// ---------------------------------------------------------------------------
__global__ void k_silu_strided(const float* __restrict__ src, float* __restrict__ dst,
                               long nPerB, long srcStride, long dstStride, int nb) {
    long total = nPerB * nb;
    long i = (long)blockIdx.x * blockDim.x + threadIdx.x;
    if (i >= total) return;
    int b = (int)(i / nPerB); long r = i % nPerB;
    dst[(long)b * dstStride + r] = siluf(src[(long)b * srcStride + r]);
}

__global__ void k_add(const float* __restrict__ a, const float* __restrict__ b,
                      float* __restrict__ o, long n) {
    long i = (long)blockIdx.x * blockDim.x + threadIdx.x;
    if (i < n) o[i] = a[i] + b[i];
}

__global__ void k_gather_dirs(const float* __restrict__ fo1, float* __restrict__ fop,
                              int B, int C, int H, int W) {
    int HW = H * W;
    long total = (long)B * HW * 4 * C;
    long i = (long)blockIdx.x * blockDim.x + threadIdx.x;
    if (i >= total) return;
    int c = (int)(i % C); long t = i / C;
    int j = (int)(t % 4); t /= 4;
    int l = (int)(t % HW); int b = (int)(t / HW);
    int sl;
    if (j == 0) sl = l;
    else if (j == 1) sl = HW - 1 - l;
    else {
        int ll = (j == 2) ? l : (HW - 1 - l);
        sl = (ll % H) * W + (ll / H);
    }
    fop[i] = fo1[((long)b * C + c) * HW + sl];
}

__global__ void k_sum4_mul(const float* __restrict__ mout, const float* __restrict__ fo2,
                           float* __restrict__ res, int B, int C, int HW) {
    long total = (long)B * C * HW;
    long i = (long)blockIdx.x * blockDim.x + threadIdx.x;
    if (i >= total) return;
    int l = (int)(i % HW); long t = i / HW;
    int c = (int)(t % C); int b = (int)(t / C);
    const float* row = mout + ((long)b * HW + l) * 4 * C + c;
    res[i] = (row[0] + row[C] + row[2 * C] + row[3 * C]) * fo2[i];
}

__global__ void k_mean_hw(const float* __restrict__ x, float* __restrict__ mean, int HW) {
    int bc = blockIdx.x;
    const float* p = x + (long)bc * HW;
    float s = 0.f;
    for (int i = threadIdx.x; i < HW; i += blockDim.x) s += p[i];
    __shared__ float sm[256];
    sm[threadIdx.x] = s; __syncthreads();
    for (int st = 128; st > 0; st >>= 1) {
        if ((int)threadIdx.x < st) sm[threadIdx.x] += sm[threadIdx.x + st];
        __syncthreads();
    }
    if (threadIdx.x == 0) mean[bc] = sm[0] / (float)HW;
}

__global__ void k_build_foc(const float* __restrict__ mean, float* __restrict__ foc2,
                            int B, int C) {
    int i = blockIdx.x * blockDim.x + threadIdx.x;
    if (i >= B * C) return;
    int l = i % C, b = i / C;
    foc2[i * 2 + 0] = mean[b * C + l];
    foc2[i * 2 + 1] = mean[b * C + (C - 1 - l)];
}

__global__ void k_chan_scale(const float* __restrict__ res, const float* __restrict__ m2o,
                             float* __restrict__ foss, int B, int C, int HW) {
    long total = (long)B * C * HW;
    long i = (long)blockIdx.x * blockDim.x + threadIdx.x;
    if (i >= total) return;
    int c = (int)((i / HW) % C); int b = (int)(i / ((long)C * HW));
    float scl = m2o[((long)b * C + c) * 2] + m2o[((long)b * C + c) * 2 + 1];
    foss[i] = res[i] * (1.f + scl);
}

__global__ void k_geglu(const float* __restrict__ f, float* __restrict__ out,
                        int B, int hid, int HW) {
    long total = (long)B * hid * HW;
    long i = (long)blockIdx.x * blockDim.x + threadIdx.x;
    if (i >= total) return;
    int l = (int)(i % HW); long t = i / HW;
    int c = (int)(t % hid); int b = (int)(t / hid);
    float v1 = f[((long)b * 2 * hid + c) * HW + l];
    float v2 = f[((long)b * 2 * hid + hid + c) * HW + l];
    out[i] = geluf(v1) * v2;
}

__global__ void k_concat_ch(const float* __restrict__ a, const float* __restrict__ b2,
                            float* __restrict__ out, int B, int C1, int C2, int HW) {
    long total = (long)B * (C1 + C2) * HW;
    long i = (long)blockIdx.x * blockDim.x + threadIdx.x;
    if (i >= total) return;
    int l = (int)(i % HW); long t = i / HW;
    int c = (int)(t % (C1 + C2)); int b = (int)(t / (C1 + C2));
    out[i] = (c < C1) ? a[((long)b * C1 + c) * HW + l]
                      : b2[((long)b * C2 + (c - C1)) * HW + l];
}

// ---------------------------------------------------------------------------
// Host-side orchestration
// ---------------------------------------------------------------------------
#define FPTR(p) ((const float*)(p))
static inline int gdim(long n) { return (int)((n + 255) / 256); }

struct Bump {
    char* base; size_t off;
    float* alloc(size_t nElems) {
        size_t bytes = (nElems * sizeof(float) + 255) & ~(size_t)255;
        float* p = (float*)(base + off); off += bytes; return p;
    }
};

struct MP {
    const float *in_w, *conv_w, *conv_b, *x_w, *dt_w, *dt_b, *A_log, *D, *out_w;
    int d_model, d_inner, dt_rank;
};
struct BP {
    const float *gn1_s, *gn1_b, *in_w, *in_b, *dw_w, *dw_b;
    MP m1, m2;
    const float *op_w, *op_b, *gn2_s, *gn2_b, *fi_w, *fi_b, *fd_w, *fd_b, *fo_w, *fo_b;
};

static MP read_mp(void* const* din, int& c, int d_model) {
    MP m; m.d_model = d_model; m.d_inner = 2 * d_model; m.dt_rank = (d_model + 15) / 16;
    m.in_w   = FPTR(din[c++]); m.conv_w = FPTR(din[c++]); m.conv_b = FPTR(din[c++]);
    m.x_w    = FPTR(din[c++]); m.dt_w   = FPTR(din[c++]); m.dt_b   = FPTR(din[c++]);
    m.A_log  = FPTR(din[c++]); m.D      = FPTR(din[c++]); m.out_w  = FPTR(din[c++]);
    return m;
}
static BP read_bp(void* const* din, int& c, int C) {
    BP p;
    p.gn1_s = FPTR(din[c++]); p.gn1_b = FPTR(din[c++]);
    p.in_w  = FPTR(din[c++]); p.in_b  = FPTR(din[c++]);
    p.dw_w  = FPTR(din[c++]); p.dw_b  = FPTR(din[c++]);
    p.m1 = read_mp(din, c, 4 * C);
    p.m2 = read_mp(din, c, 2);
    p.op_w  = FPTR(din[c++]); p.op_b  = FPTR(din[c++]);
    p.gn2_s = FPTR(din[c++]); p.gn2_b = FPTR(din[c++]);
    p.fi_w  = FPTR(din[c++]); p.fi_b  = FPTR(din[c++]);
    p.fd_w  = FPTR(din[c++]); p.fd_b  = FPTR(din[c++]);
    p.fo_w  = FPTR(din[c++]); p.fo_b  = FPTR(din[c++]);
    return p;
}

static void gemm(hipStream_t s, const float* A, const float* Wp, const float* bias, float* C,
                 int M, int N, int K, int lda, int ldb, int ldc,
                 int tB, int biasMode, int act,
                 long sA, long sB, long sC, int nb) {
    dim3 grid((N + 31) / 32, (M + 63) / 64, nb);
    if (tB == 0)
        k_gemm_t<0><<<grid, 128, 0, s>>>(A, Wp, bias, C, M, N, K, lda, ldb, ldc,
                                         biasMode, act, sA, sB, sC);
    else
        k_gemm_t<1><<<grid, 128, 0, s>>>(A, Wp, bias, C, M, N, K, lda, ldb, ldc,
                                         biasMode, act, sA, sB, sC);
}

// 1x1 conv on NCHW as per-batch GEMM: out(Cout,HW) = W(Cout,Cin) x x(Cin,HW)
static void conv1x1(hipStream_t s, const float* x, const float* w, const float* b,
                    float* out, int Cin, int Cout, int HW, int Bn) {
    gemm(s, w, x, b, out, Cout, HW, Cin, Cin, HW, HW,
         1, 2, 0, 0, (long)Cin * HW, (long)Cout * HW, Bn);
}

static void run_mamba(hipStream_t s, const float* xin, float* yout, const MP& mp,
                      int Bn, int L, Bump& ws) {
    int M = Bn * L, dm = mp.d_model, di = mp.d_inner, dr = mp.dt_rank;
    size_t mark = ws.off;
    float* xz = ws.alloc((size_t)M * 2 * di);
    gemm(s, xin, mp.in_w, nullptr, xz, M, 2 * di, dm, dm, dm, 2 * di, 0, 0, 0, 0, 0, 0, 1);
    float* u = ws.alloc((size_t)M * di);
    k_cconv1d<<<gdim((long)M * di), 256, 0, s>>>(xz, mp.conv_w, mp.conv_b, u, Bn, L, di);
    int nx = dr + 32;
    float* xd = ws.alloc((size_t)M * nx);
    gemm(s, u, mp.x_w, nullptr, xd, M, nx, di, di, di, nx, 0, 0, 0, 0, 0, 0, 1);
    float* delta = ws.alloc((size_t)M * di);
    gemm(s, xd, mp.dt_w, mp.dt_b, delta, M, di, dr, nx, dr, di, 0, 1, 2, 0, 0, 0, 1);
    float* y = ws.alloc((size_t)M * di);
    int nt = Bn * di;
    k_scan<<<(nt + 255) / 256, 256, 0, s>>>(delta, xd + dr, xd + dr + 16, u, xz + di,
                                            mp.A_log, mp.D, y, Bn, L, di, nx, 2 * di);
    gemm(s, y, mp.out_w, nullptr, yout, M, dm, di, di, di, dm, 0, 0, 0, 0, 0, 0, 1);
    ws.off = mark;
}

static void oss_block(hipStream_t s, const float* xin, float* xout, const BP& p,
                      int C, int H, int W, int groups, Bump& ws) {
    const int Bn = 2; const int HW = H * W;
    size_t mark = ws.off;
    size_t nC = (size_t)Bn * C * HW;

    float* xn = ws.alloc(nC);
    k_gn<<<dim3(groups, Bn), 256, 0, s>>>(xin, p.gn1_s, p.gn1_b, xn, C, HW, groups);
    float* f = ws.alloc(2 * nC);
    conv1x1(s, xn, p.in_w, p.in_b, f, C, 2 * C, HW, Bn);
    float* fo1 = ws.alloc(nC);
    k_dwconv3<<<gdim((long)nC), 256, 0, s>>>(f, p.dw_w, p.dw_b, fo1, Bn, C, H, W,
                                             (long)2 * C * HW, 1);
    float* fo2 = ws.alloc(nC);
    k_silu_strided<<<gdim((long)nC), 256, 0, s>>>(f + (long)C * HW, fo2,
                                                  (long)C * HW, (long)2 * C * HW,
                                                  (long)C * HW, Bn);
    float* fop = ws.alloc((size_t)Bn * HW * 4 * C);
    k_gather_dirs<<<gdim((long)Bn * HW * 4 * C), 256, 0, s>>>(fo1, fop, Bn, C, H, W);
    float* mout = ws.alloc((size_t)Bn * HW * 4 * C);
    run_mamba(s, fop, mout, p.m1, Bn, HW, ws);
    float* res = ws.alloc(nC);
    k_sum4_mul<<<gdim((long)nC), 256, 0, s>>>(mout, fo2, res, Bn, C, HW);
    float* mean = ws.alloc((size_t)Bn * C);
    k_mean_hw<<<Bn * C, 256, 0, s>>>(res, mean, HW);
    float* foc2 = ws.alloc((size_t)Bn * C * 2);
    k_build_foc<<<gdim(Bn * C), 256, 0, s>>>(mean, foc2, Bn, C);
    float* m2o = ws.alloc((size_t)Bn * C * 2);
    run_mamba(s, foc2, m2o, p.m2, Bn, C, ws);
    float* foss = ws.alloc(nC);
    k_chan_scale<<<gdim((long)nC), 256, 0, s>>>(res, m2o, foss, Bn, C, HW);
    float* xp = ws.alloc(nC);
    conv1x1(s, foss, p.op_w, p.op_b, xp, C, C, HW, Bn);
    float* x2 = ws.alloc(nC);
    k_add<<<gdim((long)nC), 256, 0, s>>>(xp, xin, x2, (long)nC);

    float* xn2 = ws.alloc(nC);
    k_gn<<<dim3(groups, Bn), 256, 0, s>>>(x2, p.gn2_s, p.gn2_b, xn2, C, HW, groups);
    float* f2 = ws.alloc(4 * nC);
    conv1x1(s, xn2, p.fi_w, p.fi_b, f2, C, 4 * C, HW, Bn);
    float* f2d = ws.alloc(4 * nC);
    k_dwconv3<<<gdim((long)(4 * nC)), 256, 0, s>>>(f2, p.fd_w, p.fd_b, f2d, Bn, 4 * C, H, W,
                                                   (long)4 * C * HW, 0);
    float* g = ws.alloc(2 * nC);
    k_geglu<<<gdim((long)(2 * nC)), 256, 0, s>>>(f2d, g, Bn, 2 * C, HW);
    float* fout = ws.alloc(nC);
    conv1x1(s, g, p.fo_w, p.fo_b, fout, 2 * C, C, HW, Bn);
    k_add<<<gdim((long)nC), 256, 0, s>>>(fout, x2, xout, (long)nC);
    ws.off = mark;
}

extern "C" void kernel_launch(void* const* d_in, const int* in_sizes, int n_in,
                              void* d_out, int out_size, void* d_ws, size_t ws_size,
                              hipStream_t stream) {
    (void)in_sizes; (void)n_in; (void)out_size; (void)ws_size;
    int c = 0;
    const float* X          = FPTR(d_in[c++]);         // (2,3,64,64)
    const float* conv_in_w  = FPTR(d_in[c++]);
    const float* conv_in_b  = FPTR(d_in[c++]);
    BP down0 = read_bp(d_in, c, 48);
    const float* ds0_w = FPTR(d_in[c++]);
    const float* ds0_b = FPTR(d_in[c++]);
    BP mid0  = read_bp(d_in, c, 96);
    const float* us0_w = FPTR(d_in[c++]);
    const float* us0_b = FPTR(d_in[c++]);
    const float* sk_w  = FPTR(d_in[c++]);
    const float* sk_b  = FPTR(d_in[c++]);
    BP up0   = read_bp(d_in, c, 48);
    BP ref0  = read_bp(d_in, c, 48);
    const float* conv_out_w = FPTR(d_in[c++]);
    const float* conv_out_b = FPTR(d_in[c++]);

    Bump ws{(char*)d_ws, 0};
    const int Bn = 2, H = 64, W = 64, HW = H * W;

    float* x1  = ws.alloc((size_t)Bn * 48 * HW);
    float* x1b = ws.alloc((size_t)Bn * 48 * HW);     // res0
    float* xd  = ws.alloc((size_t)Bn * 96 * HW / 4);
    float* xm  = ws.alloc((size_t)Bn * 96 * HW / 4);
    float* xu  = ws.alloc((size_t)Bn * 48 * HW);
    float* xc  = ws.alloc((size_t)Bn * 96 * HW);
    float* xs  = ws.alloc((size_t)Bn * 48 * HW);
    float* xs2 = ws.alloc((size_t)Bn * 48 * HW);
    float* xs3 = ws.alloc((size_t)Bn * 48 * HW);
    float* co  = ws.alloc((size_t)Bn * 3 * HW);

    k_conv2d<<<gdim((long)Bn * 48 * HW), 256, 0, stream>>>(
        X, conv_in_w, conv_in_b, x1, Bn, 3, H, W, 48, H, W, 3, 3, 1, 1);
    oss_block(stream, x1, x1b, down0, 48, H, W, 16, ws);
    k_conv2d<<<gdim((long)Bn * 96 * (HW / 4)), 256, 0, stream>>>(
        x1b, ds0_w, ds0_b, xd, Bn, 48, H, W, 96, H / 2, W / 2, 4, 4, 2, 1);
    oss_block(stream, xd, xm, mid0, 96, H / 2, W / 2, 32, ws);
    k_convT<<<gdim((long)Bn * 48 * HW), 256, 0, stream>>>(
        xm, us0_w, us0_b, xu, Bn, 96, H / 2, W / 2, 48);
    k_concat_ch<<<gdim((long)Bn * 96 * HW), 256, 0, stream>>>(xu, x1b, xc, Bn, 48, 48, HW);
    conv1x1(stream, xc, sk_w, sk_b, xs, 96, 48, HW, Bn);
    oss_block(stream, xs,  xs2, up0,  48, H, W, 16, ws);
    oss_block(stream, xs2, xs3, ref0, 48, H, W, 16, ws);
    k_conv2d<<<gdim((long)Bn * 3 * HW), 256, 0, stream>>>(
        xs3, conv_out_w, conv_out_b, co, Bn, 48, H, W, 3, H, W, 3, 3, 1, 1);
    k_add<<<gdim((long)Bn * 3 * HW), 256, 0, stream>>>(co, X, (float*)d_out, (long)Bn * 3 * HW);
}
